// MultiScaleCascadeTrans_28063316312389
// MI455X (gfx1250) — matrix operand order repaired and checked by reading, not compiled, not run on hardware
//
#include <hip/hip_runtime.h>
#include <hip/hip_bf16.h>

typedef __attribute__((ext_vector_type(16))) _Float16 v16h;
typedef __attribute__((ext_vector_type(8)))  _Float16 v8h;
typedef __attribute__((ext_vector_type(8)))  float    v8f;

#define WS    7
#define NTOK  343          // 7^3
#define NPAD  352          // 343 padded to 16*22
#define NT    22           // token tiles of 16
#define KD    16
#define HEADS 4
#define DIMC  128
#define EPSB  1e-5f
#define ATT_SCALE 0.25f    // 16^-0.5

// ---------------- dynamic LDS layout (bytes) ----------------
constexpr int FEAT_OFF = 0;                         // f32 [32][352] cascade feature / attn-out accum
constexpr int XSTG_OFF = FEAT_OFF + 32*NPAD*4;      // f32 [32][352] async-prefetched x window slice
constexpr int QF_OFF   = XSTG_OFF + 32*NPAD*4;      // f32 [16][352] q before dwconv          (head loop)
constexpr int SBUF_OFF = QF_OFF   + 16*NPAD*4;      // f32 [16][352] score tile  == featT f16 [352][32]
constexpr int PRJW_OFF = QF_OFF;                    // f16 [128][128] proj weights (staged AFTER head loop)
constexpr int ABUF_OFF = SBUF_OFF + 16*NPAD*4;      // f16 [16][352] softmaxed attention tile
constexpr int QHT_OFF  = ABUF_OFF + 16*NPAD*2;      // f16 [352][32] q^T (K-contig, ch16..31 zero)
constexpr int KHT_OFF  = QHT_OFF  + NPAD*32*2;      // f16 [352][32] k^T (K-contig, ch16..31 zero)
constexpr int VH_OFF   = KHT_OFF  + NPAD*32*2;      // f16 [32][352] v (m-contig)
constexpr int HOUT_OFF = VH_OFF   + 32*NPAD*2;      // f16 [352][128] relu(concat heads)^T (K-contig)
constexpr int QKVW_OFF = HOUT_OFF + NPAD*128*2;     // f16 [64][32]  qkv weights (per head)
constexpr int QSC_OFF  = QKVW_OFF + 64*32*2;        // f32 [64]
constexpr int QSH_OFF  = QSC_OFF  + 64*4;           // f32 [64]
constexpr int PSC_OFF  = QSH_OFF  + 64*4;           // f32 [128]
constexpr int PSH_OFF  = PSC_OFF  + 128*4;          // f32 [128]
constexpr int DWW_OFF  = PSH_OFF  + 128*4;          // f32 [16][27]
constexpr int DWSC_OFF = DWW_OFF  + 432*4;          // f32 [16]
constexpr int DWSH_OFF = DWSC_OFF + 16*4;           // f32 [16]
constexpr int SMEM_BYTES = DWSH_OFF + 16*4;         // 311,616 B <= 320 KB WGP LDS

__device__ __forceinline__ v8f wmma_f16(v16h a, v16h b, v8f c) {
  return __builtin_amdgcn_wmma_f32_16x16x32_f16(false, a, false, b, (short)0, c, false, false);
}

// A-frag: 16xK tile, row-major rows of contiguous K. lane<16: M=lane, K [0..7],[16..23];
// lane>=16: M=lane-16, K [8..15],[24..31].   (ISA 7.12.2)
__device__ __forceinline__ v16h load_frag_a(const _Float16* base, int row_stride, int lane) {
  const int hi = lane >> 4, r = lane & 15;
  const _Float16* p = base + r * row_stride + hi * 8;
  v8h x0 = *(const v8h*)(p);
  v8h x1 = *(const v8h*)(p + 16);
  v16h a;
#pragma unroll
  for (int j = 0; j < 8; ++j) { a[j] = x0[j]; a[j + 8] = x1[j]; }
  return a;
}

// B-frag: Kx16 tile stored as 16 N-major rows of contiguous K (B^T row-major).
// lane<16: N=lane, K [0..7],[16..23]; lane>=16: N=lane-16, K [8..15],[24..31].
__device__ __forceinline__ v16h load_frag_b(const _Float16* base, int row_stride, int lane) {
  const int hi = lane >> 4, n = lane & 15;
  const _Float16* p = base + n * row_stride + hi * 8;
  v8h x0 = *(const v8h*)(p);
  v8h x1 = *(const v8h*)(p + 16);
  v16h b;
#pragma unroll
  for (int j = 0; j < 8; ++j) { b[j] = x0[j]; b[j + 8] = x1[j]; }
  return b;
}

// CDNA5 async global->LDS gather (no VGPR data path, tracked by ASYNCcnt).
__device__ __forceinline__ void async_ld_b32(unsigned lds_off, const float* base, unsigned byte_off) {
  asm volatile("global_load_async_to_lds_b32 %0, %1, %2"
               :: "v"(lds_off), "v"(byte_off), "s"(base) : "memory");
}
__device__ __forceinline__ void wait_async0() {
  asm volatile("s_wait_asynccnt 0x0" ::: "memory");
}

__device__ __forceinline__ void prefetch_slice(int head, int tid, int bb, int wd, int wh, int ww,
                                               const float* x, char* smem) {
  // issue per-element async loads of this head's x window slice into XSTG
  for (int idx = tid; idx < 32 * NTOK; idx += 256) {
    int c = idx / NTOK, n = idx % NTOK;
    int dz = n / 49, r2 = n % 49, dy = r2 / 7, dx = r2 % 7;
    unsigned g = (unsigned)(((((bb * DIMC + head * 32 + c) * 28 + (wd * 7 + dz)) * 784) +
                             (wh * 7 + dy) * 28 + (ww * 7 + dx)) * 4);
    unsigned l = (unsigned)(uintptr_t)(smem + XSTG_OFF + (c * NPAD + n) * 4);
    async_ld_b32(l, x, g);
  }
}

__global__ __launch_bounds__(256)
void cascade_attn3d_kernel(const float* __restrict__ x,
                           const float* __restrict__ bias_table,
                           const float* __restrict__ qkv_w,
                           const float* __restrict__ qkv_g, const float* __restrict__ qkv_b,
                           const float* __restrict__ qkv_m, const float* __restrict__ qkv_v,
                           const float* __restrict__ dw_w,
                           const float* __restrict__ dw_g, const float* __restrict__ dw_b,
                           const float* __restrict__ dw_m, const float* __restrict__ dw_v,
                           const float* __restrict__ proj_w,
                           const float* __restrict__ proj_g, const float* __restrict__ proj_b,
                           const float* __restrict__ proj_m, const float* __restrict__ proj_v,
                           const int*   __restrict__ bias_index,
                           float* __restrict__ out) {
  extern __shared__ char smem[];
  float*    feat  = (float*)   (smem + FEAT_OFF);
  float*    xstg  = (float*)   (smem + XSTG_OFF);
  float*    qf    = (float*)   (smem + QF_OFF);
  float*    sbuf  = (float*)   (smem + SBUF_OFF);
  _Float16* featT = (_Float16*)(smem + SBUF_OFF);   // alias: live Phase A..B, dead in Phase D
  _Float16* prjw  = (_Float16*)(smem + PRJW_OFF);   // alias over qf/sbuf: staged after head loop
  _Float16* abuf  = (_Float16*)(smem + ABUF_OFF);
  _Float16* qht   = (_Float16*)(smem + QHT_OFF);
  _Float16* kht   = (_Float16*)(smem + KHT_OFF);
  _Float16* vh    = (_Float16*)(smem + VH_OFF);
  _Float16* hout  = (_Float16*)(smem + HOUT_OFF);
  _Float16* qkvw  = (_Float16*)(smem + QKVW_OFF);
  float*    qsc   = (float*)   (smem + QSC_OFF);
  float*    qsh   = (float*)   (smem + QSH_OFF);
  float*    psc   = (float*)   (smem + PSC_OFF);
  float*    psh   = (float*)   (smem + PSH_OFF);
  float*    dww   = (float*)   (smem + DWW_OFF);
  float*    dwsc  = (float*)   (smem + DWSC_OFF);
  float*    dwsh  = (float*)   (smem + DWSH_OFF);

  const int tid  = threadIdx.x;
  const int lane = tid & 31;
  const int wave = tid >> 5;

  const int wi  = blockIdx.x;           // 128 windows = B(2) * 4*4*4
  const int bb  = wi >> 6;
  const int rem = wi & 63;
  const int wd  = rem >> 4;
  const int wh  = (rem >> 2) & 3;
  const int ww  = rem & 3;

  // ---- Phase 0: kick off async prefetch of head 0 slice, stage constants, zero pads ----
  prefetch_slice(0, tid, bb, wd, wh, ww, x, smem);
  if (tid < 128) {
    float sc = proj_g[tid] * rsqrtf(proj_v[tid] + EPSB);
    psc[tid] = sc;
    psh[tid] = proj_b[tid] - proj_m[tid] * sc;
  }
  for (int idx = tid; idx < NPAD * 32; idx += 256) { qht[idx] = (_Float16)0.f; kht[idx] = (_Float16)0.f; }
  for (int idx = tid; idx < 32 * NPAD; idx += 256) { vh[idx] = (_Float16)0.f; feat[idx] = 0.f; }

  for (int head = 0; head < HEADS; ++head) {
    wait_async0();        // xstg slice for this head landed (each wave drains its own ASYNCcnt)
    __syncthreads();

    // ---- Phase A: cascade update feat += slice; build f16 K-contiguous featT for WMMA-B ----
    for (int idx = tid; idx < 32 * NPAD; idx += 256) {
      int c = idx / NPAD, n = idx % NPAD;
      float val = (n < NTOK) ? xstg[idx] : 0.f;
      float f = (head == 0) ? val : (feat[idx] + val);
      feat[idx] = f;
      featT[n * 32 + c] = (_Float16)f;
    }
    // per-head weight / BN staging
    for (int idx = tid; idx < 64 * 32; idx += 256) qkvw[idx] = (_Float16)qkv_w[head * 2048 + idx];
    if (tid < 64) {
      float sc = qkv_g[head * 64 + tid] * rsqrtf(qkv_v[head * 64 + tid] + EPSB);
      qsc[tid] = sc;
      qsh[tid] = qkv_b[head * 64 + tid] - qkv_m[head * 64 + tid] * sc;
    }
    for (int idx = tid; idx < 432; idx += 256) dww[idx] = dw_w[head * 432 + idx];
    if (tid < 16) {
      float sc = dw_g[head * 16 + tid] * rsqrtf(dw_v[head * 16 + tid] + EPSB);
      dwsc[tid] = sc;
      dwsh[tid] = dw_b[head * 16 + tid] - dw_m[head * 16 + tid] * sc;
    }
    __syncthreads();

    // overlap: prefetch next head's slice while this head computes
    if (head + 1 < HEADS) prefetch_slice(head + 1, tid, bb, wd, wh, ww, x, smem);

    // ---- Phase B: qkv 1x1x1 conv + BN, WMMA [64x32] x [32x343]; tile-uniform q/k/v routing ----
    for (int job = wave; job < 4 * NT; job += 8) {
      int ot = job / NT, nt = job % NT;
      int o0 = ot * 16, n0 = nt * 16;
      v16h a = load_frag_a(qkvw + o0 * 32, 32, lane);
      v16h bf = load_frag_b(featT + n0 * 32, 32, lane);
      v8f c = {};
      c = wmma_f16(a, bf, c);
      int hi = lane >> 4, nn = n0 + (lane & 15);
      if (nn < NTOK) {
        if (ot == 0) {                       // q rows (f32, pre-dwconv)
#pragma unroll
          for (int r = 0; r < 8; ++r) {
            int o = r + 8 * hi;
            qf[o * NPAD + nn] = c[r] * qsc[o] + qsh[o];
          }
        } else if (ot == 1) {                // k^T: 8 consecutive channels -> one b128 store
          v8h pack;
#pragma unroll
          for (int r = 0; r < 8; ++r) {
            int o = 16 + r + 8 * hi;
            pack[r] = (_Float16)(c[r] * qsc[o] + qsh[o]);
          }
          *(v8h*)(kht + nn * 32 + 8 * hi) = pack;
        } else {                             // v rows
          int d0 = (ot - 2) * 16 + 8 * hi;
#pragma unroll
          for (int r = 0; r < 8; ++r) {
            int o = 32 + d0 + r;
            vh[(d0 + r) * NPAD + nn] = (_Float16)(c[r] * qsc[o] + qsh[o]);
          }
        }
      }
    }
    __syncthreads();

    // ---- Phase C: depthwise 3x3x3 conv + BN on q, fold attn scale, write q^T f16 ----
    for (int idx = tid; idx < KD * NTOK; idx += 256) {
      int c = idx / NTOK, n = idx % NTOK;
      int dz = n / 49, r2 = n % 49, dy = r2 / 7, dx = r2 % 7;
      float acc = 0.f;
#pragma unroll
      for (int kz = 0; kz < 3; ++kz) {
        int z = dz + kz - 1;
        if (z < 0 || z >= WS) continue;
#pragma unroll
        for (int ky = 0; ky < 3; ++ky) {
          int y = dy + ky - 1;
          if (y < 0 || y >= WS) continue;
#pragma unroll
          for (int kx = 0; kx < 3; ++kx) {
            int xx = dx + kx - 1;
            if (xx < 0 || xx >= WS) continue;
            acc += qf[c * NPAD + (z * 7 + y) * 7 + xx] * dww[c * 27 + (kz * 3 + ky) * 3 + kx];
          }
        }
      }
      qht[n * 32 + c] = (_Float16)((acc * dwsc[c] + dwsh[c]) * ATT_SCALE);
    }
    for (int idx = tid; idx < 32 * NPAD; idx += 256) feat[idx] = 0.f;  // -> attn-out accumulator
    __syncthreads();

    // ---- Phase D: attention, one 16-query tile at a time ----
    for (int qt = 0; qt < NT; ++qt) {
      v16h aq = load_frag_a(qht + qt * 16 * 32, 32, lane);
      // scores S = Q^T K + bias : 22 key tiles over 8 waves
      for (int mt = wave; mt < NT; mt += 8) {
        v16h bk = load_frag_b(kht + mt * 16 * 32, 32, lane);
        v8f c = {};
        c = wmma_f16(aq, bk, c);
        int hi = lane >> 4;
        int m = mt * 16 + (lane & 15);
        if (mt < 21 && qt < 21) {            // interior: branch-free
#pragma unroll
          for (int r = 0; r < 8; ++r) {
            int nl = r + 8 * hi;
            int n = qt * 16 + nl;
            sbuf[nl * NPAD + m] = c[r] + bias_table[bias_index[n * NTOK + m] * HEADS + head];
          }
        } else {                             // edge tiles
#pragma unroll
          for (int r = 0; r < 8; ++r) {
            int nl = r + 8 * hi;
            int n = qt * 16 + nl;
            float s;
            if (m < NTOK && n < NTOK) s = c[r] + bias_table[bias_index[n * NTOK + m] * HEADS + head];
            else                      s = (m >= NTOK) ? -1e30f : 0.f;
            sbuf[nl * NPAD + m] = s;
          }
        }
      }
      __syncthreads();
      // softmax over keys: each wave owns 2 rows
#pragma unroll
      for (int rr = 0; rr < 2; ++rr) {
        int row = wave * 2 + rr;
        float mx = -1e30f;
        for (int j = lane; j < NPAD; j += 32) mx = fmaxf(mx, sbuf[row * NPAD + j]);
#pragma unroll
        for (int s = 16; s > 0; s >>= 1) mx = fmaxf(mx, __shfl_xor(mx, s, 32));
        float sum = 0.f;
        for (int j = lane; j < NPAD; j += 32) {
          float e = __expf(sbuf[row * NPAD + j] - mx);
          sum += e;
          abuf[row * NPAD + j] = (_Float16)e;
        }
#pragma unroll
        for (int s = 16; s > 0; s >>= 1) sum += __shfl_xor(sum, s, 32);
        float inv = 1.0f / sum;
        for (int j = lane; j < NPAD; j += 32)
          abuf[row * NPAD + j] = (_Float16)((float)abuf[row * NPAD + j] * inv);
      }
      __syncthreads();
      // A * V^T over K=352 (11 chunks of 32) split across waves; LDS f32 atomics reduce
      {
        v8f c0 = {}, c1 = {};
        for (int ks = wave; ks < 11; ks += 8) {
          v16h aa = load_frag_a(abuf + ks * 32, NPAD, lane);
          v16h b0 = load_frag_b(vh + ks * 32, NPAD, lane);
          v16h b1 = load_frag_b(vh + 16 * NPAD + ks * 32, NPAD, lane);
          c0 = wmma_f16(aa, b0, c0);
          c1 = wmma_f16(aa, b1, c1);
        }
        int hi = lane >> 4;
        int d0 = lane & 15;
        if (qt < 21) {
#pragma unroll
          for (int r = 0; r < 8; ++r) {
            int n = qt * 16 + r + 8 * hi;
            atomicAdd(&feat[d0 * NPAD + n], c0[r]);
            atomicAdd(&feat[(16 + d0) * NPAD + n], c1[r]);
          }
        } else {
#pragma unroll
          for (int r = 0; r < 8; ++r) {
            int n = qt * 16 + r + 8 * hi;
            if (n < NTOK) {
              atomicAdd(&feat[d0 * NPAD + n], c0[r]);
              atomicAdd(&feat[(16 + d0) * NPAD + n], c1[r]);
            }
          }
        }
      }
      __syncthreads();
    }

    // ---- Phase E: relu(head out) -> hout^T (packed b128 stores); feat keeps pre-relu cascade ----
    for (int idx = tid; idx < NTOK * 4; idx += 256) {
      int n = idx >> 2, g = idx & 3;
      int c0 = g * 8;
      v8h pack;
#pragma unroll
      for (int j = 0; j < 8; ++j) pack[j] = (_Float16)fmaxf(feat[(c0 + j) * NPAD + n], 0.f);
      *(v8h*)(hout + n * DIMC + head * 32 + c0) = pack;
    }
    __syncthreads();
  }

  // ---- stage proj weights (aliases dead qf/sbuf region) ----
  for (int idx = tid; idx < DIMC * DIMC; idx += 256) prjw[idx] = (_Float16)proj_w[idx];
  __syncthreads();

  // ---- Phase F: proj GEMM [128x128] x [128x343] + BN + window-reverse scatter ----
  for (int job = wave; job < 8 * NT; job += 8) {
    int ot = job / NT, nt = job % NT;
    v8f acc = {};
#pragma unroll
    for (int ks = 0; ks < 4; ++ks) {
      v16h a = load_frag_a(prjw + ot * 16 * DIMC + ks * 32, DIMC, lane);
      v16h bb2 = load_frag_b(hout + nt * 16 * DIMC + ks * 32, DIMC, lane);
      acc = wmma_f16(a, bb2, acc);
    }
    int hi = lane >> 4;
    int n = nt * 16 + (lane & 15);
    if (nt < 21 || n < NTOK) {
      int dz = n / 49, r2 = n % 49, dy = r2 / 7, dx = r2 % 7;
      int zz = wd * 7 + dz, yy = wh * 7 + dy, xx = ww * 7 + dx;
#pragma unroll
      for (int r = 0; r < 8; ++r) {
        int o = ot * 16 + r + 8 * hi;
        out[((bb * DIMC + o) * 28 + zz) * 784 + yy * 28 + xx] = acc[r] * psc[o] + psh[o];
      }
    }
  }
}

extern "C" void kernel_launch(void* const* d_in, const int* in_sizes, int n_in,
                              void* d_out, int out_size, void* d_ws, size_t ws_size,
                              hipStream_t stream) {
  const float* x          = (const float*)d_in[0];
  const float* bias_table = (const float*)d_in[1];
  const float* qkv_w      = (const float*)d_in[2];
  const float* qkv_g      = (const float*)d_in[3];
  const float* qkv_b      = (const float*)d_in[4];
  const float* qkv_m      = (const float*)d_in[5];
  const float* qkv_v      = (const float*)d_in[6];
  const float* dw_w       = (const float*)d_in[7];
  const float* dw_g       = (const float*)d_in[8];
  const float* dw_b       = (const float*)d_in[9];
  const float* dw_m       = (const float*)d_in[10];
  const float* dw_v       = (const float*)d_in[11];
  const float* proj_w     = (const float*)d_in[12];
  const float* proj_g     = (const float*)d_in[13];
  const float* proj_b     = (const float*)d_in[14];
  const float* proj_m     = (const float*)d_in[15];
  const float* proj_v     = (const float*)d_in[16];
  const int*   bias_index = (const int*)  d_in[17];
  float* out = (float*)d_out;

  (void)hipFuncSetAttribute((const void*)cascade_attn3d_kernel,
                            hipFuncAttributeMaxDynamicSharedMemorySize, SMEM_BYTES);

  cascade_attn3d_kernel<<<dim3(128), dim3(256), SMEM_BYTES, stream>>>(
      x, bias_table, qkv_w, qkv_g, qkv_b, qkv_m, qkv_v,
      dw_w, dw_g, dw_b, dw_m, dw_v,
      proj_w, proj_g, proj_b, proj_m, proj_v, bias_index, out);
}
